// ViTEncoder_FSM_38147899523214
// MI455X (gfx1250) — matrix operand order repaired and checked by reading, not compiled, not run on hardware
//
#include <hip/hip_runtime.h>
#include <hip/hip_fp16.h>

// ---------------------------------------------------------------------------
// DynamicViT forward for MI455X (gfx1250): all GEMMs via V_WMMA_F32_16X16X32_F16,
// LDS staging via GLOBAL_LOAD_ASYNC_TO_LDS_B128 where the toolchain exposes it.
// ---------------------------------------------------------------------------

typedef __attribute__((ext_vector_type(16))) _Float16 v16h;
typedef __attribute__((ext_vector_type(8)))  float    v8f;
typedef int v4i __attribute__((vector_size(16)));

#if defined(__AMDGCN__) && __has_builtin(__builtin_amdgcn_global_load_async_to_lds_b128)
#define ASYNC_OK 1
#else
#define ASYNC_OK 0
#endif

#define GAS1 __attribute__((address_space(1)))
#define LAS3 __attribute__((address_space(3)))

union FragU { uint4 q[2]; v16h v; };

__device__ __forceinline__ v8f wmma_f16(v16h a, v16h b, v8f c) {
  return __builtin_amdgcn_wmma_f32_16x16x32_f16(false, a, false, b, (short)0, c, false, false);
}

#if ASYNC_OK
// 16-byte async DMA global -> LDS (per-lane), tracked by ASYNCcnt.
__device__ __forceinline__ void async_cp16(const __half* gp, __half* lp) {
  __builtin_amdgcn_global_load_async_to_lds_b128(
      (GAS1 v4i*)(uintptr_t)gp,
      (LAS3 v4i*)(unsigned)(uintptr_t)lp, 0, 0);
}
#endif

template <int N>
__device__ __forceinline__ void wait_async_le() {
#if defined(__AMDGCN__) && __has_builtin(__builtin_amdgcn_s_wait_asynccnt)
  __builtin_amdgcn_s_wait_asynccnt((short)N);
#elif defined(__AMDGCN__)
  asm volatile("s_wait_asynccnt %0" ::"i"(N) : "memory");
#endif
}

// A-matrix 16x32 f16 fragment (ISA 7.12.2): lanes 0-15 hold K=0..7,16..23 of row M=lane;
// lanes 16-31 hold K=8..15,24..31.
__device__ __forceinline__ v16h load_frag_a(const __half* base, int ld, int lane) {
  int r = lane & 15, hi = lane >> 4;
  const __half* p = base + (size_t)r * ld + hi * 8;
  FragU f;
  f.q[0] = *reinterpret_cast<const uint4*>(p);
  f.q[1] = *reinterpret_cast<const uint4*>(p + 16);
  return f.v;
}

// B-matrix 32x16 f16 fragment: lanes 0-15 hold K=0..7,16..23 of column N=lane; lanes 16-31
// hold K=8..15,24..31.  Source is Bt (row n = column of B), row-major.
__device__ __forceinline__ v16h load_frag_b(const __half* base, int ld, int lane) {
  int r = lane & 15, hi = lane >> 4;
  const __half* p = base + (size_t)r * ld + hi * 8;
  FragU f;
  f.q[0] = *reinterpret_cast<const uint4*>(p);
  f.q[1] = *reinterpret_cast<const uint4*>(p + 16);
  return f.v;
}

__device__ __forceinline__ float gelu_exact(float x) {
  return 0.5f * x * (1.0f + erff(x * 0.70710678118654752f));
}

// ---------------------------------------------------------------------------
// Generic GEMM: C[M,N] = A[M,K](f16) * Bt[N,K]^T(f16)  (+bias)(gelu)(+res)
// flags: 1=bias, 2=gelu, 4=residual(f32), 8=output f16 (else f32)
// M%16==0, N%BN==0, K%32==0.  Tile BMxBNx32, 256 threads (8 waves),
// double-buffered LDS, async global->LDS staging when available.
// ---------------------------------------------------------------------------
template <int BN>
__global__ __launch_bounds__(256)
void gemm_kernel(const __half* __restrict__ A, const __half* __restrict__ Bt,
                 const float* __restrict__ bias, const float* __restrict__ res,
                 void* __restrict__ out, int M, int N, int K, int flags) {
  constexpr int BM = 128;
  constexpr int WGN = (BN == 128) ? 4 : 2;   // waves along N
  constexpr int WGM = 8 / WGN;               // waves along M
  constexpr int WM = BM / WGM;               // 64 or 32
  constexpr int MT = WM / 16;                // 4 or 2 m-tiles per wave
  constexpr int PER_STAGE = (BN == 128) ? 4 : 3;  // async ops per thread per stage

  __shared__ __align__(16) __half sA[2][BM * 32];
  __shared__ __align__(16) __half sB[2][BN * 32];

  int t = threadIdx.x;
  int lane = t & 31, wid = t >> 5;
  int wm = wid / WGN, wn = wid % WGN;
  int m0 = blockIdx.y * BM, n0 = blockIdx.x * BN;
  int steps = K >> 5;

  // per-thread staging assignment
  int ar = t >> 1, aseg = t & 1;                     // A: 32B per thread
  int arow = m0 + ar; if (arow >= M) arow = M - 1;
  const __half* agp = A + (size_t)arow * K + aseg * 16;
  __half* alp0 = &sA[0][ar * 32 + aseg * 16];
  __half* alp1 = &sA[1][ar * 32 + aseg * 16];

  int br = (BN == 128) ? (t >> 1) : (t >> 2);
  int bseg = (BN == 128) ? (t & 1) : (t & 3);
  int bsh = (BN == 128) ? (bseg * 16) : (bseg * 8);  // halves within 32-wide row
  const __half* bgp = Bt + (size_t)(n0 + br) * K + bsh;
  __half* blp0 = &sB[0][br * 32 + bsh];
  __half* blp1 = &sB[1][br * 32 + bsh];

  v8f acc[MT][2] = {};

#if ASYNC_OK
  auto stage_async = [&](int kt, int buf) {
    const __half* ag = agp + kt * 32;
    __half* al = buf ? alp1 : alp0;
    async_cp16(ag, al);
    async_cp16(ag + 8, al + 8);
    const __half* bg = bgp + kt * 32;
    __half* bl = buf ? blp1 : blp0;
    async_cp16(bg, bl);
    if (BN == 128) async_cp16(bg + 8, bl + 8);
  };
  stage_async(0, 0);
#else
  uint4 ra0, ra1, rb0, rb1;
  auto gload = [&](int kt) {
    const uint4* ag = reinterpret_cast<const uint4*>(agp + kt * 32);
    ra0 = ag[0]; ra1 = ag[1];
    const uint4* bg = reinterpret_cast<const uint4*>(bgp + kt * 32);
    rb0 = bg[0];
    if (BN == 128) rb1 = bg[1];
  };
  gload(0);
#endif

  for (int kt = 0; kt < steps; kt++) {
    int buf = kt & 1;
#if ASYNC_OK
    if (kt + 1 < steps) {
      stage_async(kt + 1, buf ^ 1);
      wait_async_le<PER_STAGE>();
    } else {
      wait_async_le<0>();
    }
#else
    {
      uint4* ad = reinterpret_cast<uint4*>(buf ? alp1 : alp0);
      ad[0] = ra0; ad[1] = ra1;
      uint4* bd = reinterpret_cast<uint4*>(buf ? blp1 : blp0);
      bd[0] = rb0;
      if (BN == 128) bd[1] = rb1;
    }
    if (kt + 1 < steps) gload(kt + 1);
#endif
    if (kt + 2 < steps) {                     // L2 prefetch (global_prefetch_b8)
      __builtin_prefetch(agp + (kt + 2) * 32, 0, 1);
      __builtin_prefetch(bgp + (kt + 2) * 32, 0, 1);
    }
    __syncthreads();

    v16h af[MT], bf[2];
    for (int mi = 0; mi < MT; mi++)
      af[mi] = load_frag_a(&sA[buf][(wm * WM + mi * 16) * 32], 32, lane);
    for (int ni = 0; ni < 2; ni++)
      bf[ni] = load_frag_b(&sB[buf][(wn * 32 + ni * 16) * 32], 32, lane);
    for (int mi = 0; mi < MT; mi++)
      for (int ni = 0; ni < 2; ni++)
        acc[mi][ni] = wmma_f16(af[mi], bf[ni], acc[mi][ni]);
    __syncthreads();
  }

  int r15 = lane & 15, hi = lane >> 4;
  for (int mi = 0; mi < MT; mi++) {
    for (int ni = 0; ni < 2; ni++) {
      int colg = n0 + wn * 32 + ni * 16 + r15;
      float bv = (flags & 1) ? bias[colg] : 0.f;
      for (int rr = 0; rr < 8; rr++) {
        int rowg = m0 + wm * WM + mi * 16 + rr + hi * 8;
        if (rowg < M) {
          float v = acc[mi][ni][rr] + bv;
          if (flags & 2) v = gelu_exact(v);
          if (flags & 4) v += res[(size_t)rowg * N + colg];
          if (flags & 8) reinterpret_cast<__half*>(out)[(size_t)rowg * N + colg] = __float2half(v);
          else           reinterpret_cast<float*>(out)[(size_t)rowg * N + colg] = v;
        }
      }
    }
  }
}

// ---------------------------------------------------------------------------
// Fused attention: per (16-row tile, head, batch). N=577 (37 tiles of 16).
// qkv layout: row = b*577 + n (padded buffer), cols: q[0:768] k[768:1536] v[1536:2304].
// ---------------------------------------------------------------------------
#define ATT_SW 608 /* padded score width (37*16=592 -> 608) */

__global__ __launch_bounds__(256)
void attn_kernel(const __half* __restrict__ qkv, const float* __restrict__ pol,
                 __half* __restrict__ out, int use_policy) {
  __shared__ __align__(16) float  sS[16 * ATT_SW];
  __shared__ __align__(16) __half sP[16 * ATT_SW];
  __shared__ __align__(16) __half sVt[2][64 * 32];
  __shared__ float sRed[256];
  __shared__ float sRmax[16], sRsum[16];

  int t = threadIdx.x;
  int lane = t & 31, wid = t >> 5;
  int i0 = blockIdx.x * 16;
  int h  = blockIdx.y;
  int b  = blockIdx.z;
  int r15 = lane & 15, hi = lane >> 4;

  // ---- phase 1: S = q @ k^T * scale ------------------------------------
  const __half* qb = qkv + (size_t)(b * 577 + i0) * 2304 + h * 64;
  v16h qa0 = load_frag_a(qb, 2304, lane);
  v16h qa1 = load_frag_a(qb + 32, 2304, lane);
  for (int j = wid; j < 37; j += 8) {
    const __half* kb = qkv + (size_t)(b * 577 + j * 16) * 2304 + 768 + h * 64;
    v16h b0 = load_frag_b(kb, 2304, lane);
    v16h b1 = load_frag_b(kb + 32, 2304, lane);
    v8f a = {};
    a = wmma_f16(qa0, b0, a);
    a = wmma_f16(qa1, b1, a);
    for (int rr = 0; rr < 8; rr++)
      sS[(rr + hi * 8) * ATT_SW + j * 16 + r15] = a[rr] * 0.125f;
  }
  __syncthreads();

  // ---- phase 2: (policy-)softmax over 577 valid columns ----------------
  {
    int row = t >> 4, sub = t & 15;
    int gi = i0 + row;
    float mx = -1e30f;
    for (int c = sub; c < 577; c += 16) mx = fmaxf(mx, sS[row * ATT_SW + c]);
    sRed[t] = mx;
    __syncthreads();
    if (sub == 0) {
      float m2 = -1e30f;
      for (int i = 0; i < 16; i++) m2 = fmaxf(m2, sRed[row * 16 + i]);
      sRmax[row] = m2;
    }
    __syncthreads();
    float rm = sRmax[row];
    float sum = 0.f;
    for (int c = sub; c < 577; c += 16) {
      float e = expf(sS[row * ATT_SW + c] - rm);
      if (use_policy) {
        float ap = (c == gi) ? 1.f : pol[b * 577 + c];
        e *= ap;
      }
      sS[row * ATT_SW + c] = e;
      sum += e;
    }
    sRed[t] = sum;
    __syncthreads();
    if (sub == 0) {
      float s2 = 0.f;
      for (int i = 0; i < 16; i++) s2 += sRed[row * 16 + i];
      sRsum[row] = s2;
    }
    __syncthreads();
    float rs = sRsum[row];
    for (int c = sub; c < ATT_SW; c += 16) {
      float p = 0.f;
      if (c < 577) {
        float e = sS[row * ATT_SW + c];
        p = use_policy ? (e + 1e-6f / 577.f) / (rs + 1e-6f) : e / rs;
      }
      sP[row * ATT_SW + c] = __float2half(p);
    }
  }
  __syncthreads();

  // ---- phase 3: O = P @ V, split-K over two wave groups ------------------
  v8f accO = {};
  int grp = wid >> 2;    // 0: even K-chunks, 1: odd K-chunks
  int nt  = wid & 3;     // 16-wide n-tile within head dim
  for (int it = 0; it < (ATT_SW / 64 + 1); it++) {   // 10 iterations, chunks 2it, 2it+1
    __syncthreads();
    for (int idx = t; idx < 4096; idx += 256) {
      int bufi = idx >> 11;
      int e = idx & 2047;
      int n = e & 63, kk = e >> 6;   // coalesced over n
      int kc = 2 * it + bufi;
      if (kc < ATT_SW / 32)
        sVt[bufi][n * 32 + kk] =
            qkv[(size_t)(b * 577 + kc * 32 + kk) * 2304 + 1536 + h * 64 + n];
    }
    __syncthreads();
    int kc = 2 * it + grp;
    if (kc < ATT_SW / 32) {
      v16h pa = load_frag_a(sP + kc * 32, ATT_SW, lane);
      v16h vb = load_frag_b(&sVt[grp][nt * 16 * 32], 32, lane);
      accO = wmma_f16(pa, vb, accO);
    }
  }
  // reduce odd-group partials into even group via LDS (reuse sS)
  __syncthreads();
  if (grp == 1) {
    for (int rr = 0; rr < 8; rr++)
      sS[(nt * 16 + rr + hi * 8) * 16 + r15] = accO[rr];
  }
  __syncthreads();
  if (grp == 0) {
    for (int rr = 0; rr < 8; rr++) {
      float v = accO[rr] + sS[(nt * 16 + rr + hi * 8) * 16 + r15];
      int n = i0 + rr + hi * 8;
      if (n < 577)
        out[(size_t)(b * 577 + n) * 768 + h * 64 + nt * 16 + r15] = __float2half(v);
    }
  }
}

// ---------------------------------------------------------------------------
// LayerNorm over D=768. One block per row. flags: 1=f16 output, 2=skip-cls
// row remap (row -> b*577 + 1 + t for fsm input).
// ---------------------------------------------------------------------------
__global__ __launch_bounds__(256)
void ln_kernel(const float* __restrict__ in, const float* __restrict__ s,
               const float* __restrict__ bsh, void* __restrict__ out, int flags) {
  __shared__ float red[256];
  int row = blockIdx.x;
  int in_row = (flags & 2) ? (row + row / 576 + 1) : row;
  const float* x = in + (size_t)in_row * 768;
  int t = threadIdx.x;
  float v0 = x[t], v1 = x[t + 256], v2 = x[t + 512];
  red[t] = v0 + v1 + v2;
  __syncthreads();
  for (int st = 128; st > 0; st >>= 1) { if (t < st) red[t] += red[t + st]; __syncthreads(); }
  float mean = red[0] / 768.f;
  __syncthreads();
  float d0 = v0 - mean, d1 = v1 - mean, d2 = v2 - mean;
  red[t] = d0 * d0 + d1 * d1 + d2 * d2;
  __syncthreads();
  for (int st = 128; st > 0; st >>= 1) { if (t < st) red[t] += red[t + st]; __syncthreads(); }
  float rstd = rsqrtf(red[0] / 768.f + 1e-5f);
  float y0 = d0 * rstd * s[t]       + bsh[t];
  float y1 = d1 * rstd * s[t + 256] + bsh[t + 256];
  float y2 = d2 * rstd * s[t + 512] + bsh[t + 512];
  if (flags & 1) {
    __half* o = reinterpret_cast<__half*>(out) + (size_t)row * 768;
    o[t] = __float2half(y0); o[t + 256] = __float2half(y1); o[t + 512] = __float2half(y2);
  } else {
    float* o = reinterpret_cast<float*>(out) + (size_t)row * 768;
    o[t] = y0; o[t + 256] = y1; o[t + 512] = y2;
  }
}

// ---------------------------------------------------------------------------
// Small utility kernels
// ---------------------------------------------------------------------------
__global__ void k_convert(const float* __restrict__ in, __half* __restrict__ out, size_t n) {
  size_t i = (size_t)blockIdx.x * 256 + threadIdx.x;
  if (i < n) out[i] = __float2half(in[i]);
}

// out[n*K + k] = in[k*N + n]
__global__ void k_transpose_convert(const float* __restrict__ in, __half* __restrict__ out,
                                    int K, int N) {
  size_t i = (size_t)blockIdx.x * 256 + threadIdx.x;
  size_t total = (size_t)K * N;
  if (i < total) {
    int k = (int)(i % K), n = (int)(i / K);
    out[i] = __float2half(in[(size_t)k * N + n]);
  }
}

__global__ void k_fill1(float* p, int n) {
  int i = blockIdx.x * 256 + threadIdx.x;
  if (i < n) p[i] = 1.0f;
}

// Gather patches: A[b*576 + hh*24 + ww][c*256 + p*16 + q] = x[b,c,hh*16+p,ww*16+q]
__global__ void k_patch_gather(const float* __restrict__ x, __half* __restrict__ A) {
  size_t i = (size_t)blockIdx.x * 256 + threadIdx.x;
  if (i >= (size_t)9216 * 768) return;
  int col = (int)(i % 768), row = (int)(i / 768);
  int b = row / 576, tt = row % 576;
  int hh = tt / 24, ww = tt % 24;
  int c = col >> 8, rem = col & 255, p = rem >> 4, q = rem & 15;
  size_t xi = (((size_t)(b * 3 + c) * 384) + hh * 16 + p) * 384 + ww * 16 + q;
  A[i] = __float2half(x[xi]);
}

// tok[b,0,:] = cls + pos[0]; tok[b,1+t,:] = tmp[b*576+t] + pos[1+t]
__global__ void k_assemble(const float* __restrict__ tmp, const float* __restrict__ cls,
                           const float* __restrict__ pos, float* __restrict__ tok) {
  size_t i = (size_t)blockIdx.x * 256 + threadIdx.x;
  if (i >= (size_t)16 * 577 * 768) return;
  int d = (int)(i % 768), row = (int)(i / 768);
  int b = row / 577, n = row % 577;
  float pe = pos[(size_t)n * 768 + d];
  float v = (n == 0) ? cls[d] : tmp[((size_t)b * 576 + (n - 1)) * 768 + d];
  tok[i] = v + pe;
}

// glb[b,c] = sum_n h[b,n,384+c]*pm[b,n] / sum_n pm[b,n]   (blockDim=384)
__global__ void k_glb(const float* __restrict__ h, const float* __restrict__ pm,
                      float* __restrict__ glb) {
  __shared__ float red[384];
  int b = blockIdx.x, t = threadIdx.x;
  float ps = 0.f;
  for (int n = t; n < 576; n += 384) ps += pm[b * 576 + n];
  red[t] = ps;
  __syncthreads();
  if (t == 0) { float s = 0.f; for (int i = 0; i < 384; i++) s += red[i]; red[0] = s; }
  __syncthreads();
  float msum = red[0];
  float acc = 0.f;
  for (int n = 0; n < 576; n++)
    acc += h[((size_t)(b * 576 + n)) * 768 + 384 + t] * pm[b * 576 + n];
  glb[b * 384 + t] = acc / msum;
}

// feat[r, c] = c<384 ? h[r,c] : c<768 ? glb[b, c-384] : noise[c-768]   (f16)
__global__ void k_feat(const float* __restrict__ h, const float* __restrict__ glb,
                       const float* __restrict__ noise, __half* __restrict__ feat) {
  size_t i = (size_t)blockIdx.x * 256 + threadIdx.x;
  if (i >= (size_t)9216 * 1152) return;
  int c = (int)(i % 1152), r = (int)(i / 1152);
  int b = r / 576;
  float v;
  if (c < 384)      v = h[(size_t)r * 768 + c];
  else if (c < 768) v = glb[b * 384 + (c - 384)];
  else              v = noise[c - 768];
  feat[i] = __float2half(v);
}

// o3 head + gumbel straight-through argmax; pm[r] *= (argmax==0)
__global__ void k_o3(const __half* __restrict__ z2, const float* __restrict__ w3,
                     const float* __restrict__ b3, const float* __restrict__ gn,
                     float* __restrict__ pm) {
  int r = blockIdx.x * 256 + threadIdx.x;
  if (r >= 9216) return;
  float l0 = b3[0], l1 = b3[1];
  for (int k = 0; k < 192; k++) {
    float z = __half2float(z2[(size_t)r * 192 + k]);
    l0 += z * w3[2 * k]; l1 += z * w3[2 * k + 1];
  }
  float mx = fmaxf(l0, l1);
  float lse = mx + logf(expf(l0 - mx) + expf(l1 - mx));
  float s0 = (l0 - lse) + gn[(size_t)r * 2];
  float s1 = (l1 - lse) + gn[(size_t)r * 2 + 1];
  if (!(s0 >= s1)) pm[r] = 0.f;
}

__global__ void k_policy(const float* __restrict__ pm, float* __restrict__ pol) {
  int i = blockIdx.x * 256 + threadIdx.x;
  if (i >= 9232) return;
  int n = i % 577;
  pol[i] = (n == 0) ? 1.f : pm[(i / 577) * 576 + n - 1];
}

// ---------------------------------------------------------------------------
// Host orchestration
// ---------------------------------------------------------------------------
extern "C" void kernel_launch(void* const* d_in, const int* in_sizes, int n_in,
                              void* d_out, int out_size, void* d_ws, size_t ws_size,
                              hipStream_t stream) {
  (void)in_sizes; (void)n_in; (void)out_size; (void)ws_size;
  const float* x        = (const float*)d_in[0];
  const float* patch_w  = (const float*)d_in[1];
  const float* patch_b  = (const float*)d_in[2];
  const float* cls      = (const float*)d_in[3];
  const float* pos      = (const float*)d_in[4];
  const float* ln1_s    = (const float*)d_in[5];
  const float* ln1_b    = (const float*)d_in[6];
  const float* qkv_w    = (const float*)d_in[7];
  const float* proj_w   = (const float*)d_in[8];
  const float* proj_b   = (const float*)d_in[9];
  const float* ln2_s    = (const float*)d_in[10];
  const float* ln2_b    = (const float*)d_in[11];
  const float* fc1_w    = (const float*)d_in[12];
  const float* fc1_b    = (const float*)d_in[13];
  const float* fc2_w    = (const float*)d_in[14];
  const float* fc2_b    = (const float*)d_in[15];
  const float* fsm_ln_s = (const float*)d_in[16];
  const float* fsm_ln_b = (const float*)d_in[17];
  const float* fsm_in_w = (const float*)d_in[18];
  const float* fsm_in_b = (const float*)d_in[19];
  const float* o1_w     = (const float*)d_in[20];
  const float* o1_b     = (const float*)d_in[21];
  const float* o2_w     = (const float*)d_in[22];
  const float* o2_b     = (const float*)d_in[23];
  const float* o3_w     = (const float*)d_in[24];
  const float* o3_b     = (const float*)d_in[25];
  const float* noise    = (const float*)d_in[26];
  const float* gnoise   = (const float*)d_in[27];
  const float* norm_s   = (const float*)d_in[28];
  const float* norm_b   = (const float*)d_in[29];

  char* wsp = (char*)d_ws;
  size_t off = 0;
  auto alloc = [&](size_t bytes) -> char* {
    char* p = wsp + off;
    off += bytes;
    off = (off + 255) & ~(size_t)255;
    return p;
  };

  // f16 transposed/converted weights
  __half* patchW16 = (__half*)alloc((size_t)768 * 768 * 2);
  __half* qkvT = (__half*)alloc((size_t)12 * 2304 * 768 * 2);
  __half* projT = (__half*)alloc((size_t)12 * 768 * 768 * 2);
  __half* fc1T  = (__half*)alloc((size_t)12 * 3072 * 768 * 2);
  __half* fc2T  = (__half*)alloc((size_t)12 * 768 * 3072 * 2);
  __half* fsmT  = (__half*)alloc((size_t)2 * 768 * 768 * 2);
  __half* o1T   = (__half*)alloc((size_t)2 * 384 * 1152 * 2);
  __half* o2T   = (__half*)alloc((size_t)2 * 192 * 384 * 2);
  // f16 activations
  __half* patchA  = (__half*)alloc((size_t)9216 * 768 * 2);
  __half* lnbuf   = (__half*)alloc((size_t)9232 * 768 * 2);
  __half* qkvbuf  = (__half*)alloc((size_t)9280 * 2304 * 2);  // 48 pad rows for tile reads
  __half* attnout = (__half*)alloc((size_t)9232 * 768 * 2);
  __half* fc1out  = (__half*)alloc((size_t)9232 * 3072 * 2);
  __half* featb   = (__half*)alloc((size_t)9216 * 1152 * 2);
  __half* z1b     = (__half*)alloc((size_t)9216 * 384 * 2);
  __half* z2b     = (__half*)alloc((size_t)9216 * 192 * 2);
  // f32 buffers
  float* tok  = (float*)alloc((size_t)9232 * 768 * 4);
  float* hbuf = (float*)alloc((size_t)9232 * 768 * 4);
  float* hfsm = (float*)alloc((size_t)9216 * 768 * 4);
  float* glb  = (float*)alloc((size_t)16 * 384 * 4);
  float* pm   = (float*)alloc((size_t)9216 * 4);
  float* pol  = (float*)alloc((size_t)9232 * 4);

  auto blocks = [](size_t n) { return dim3((unsigned)((n + 255) / 256)); };
  auto gemm = [&](const __half* A, const __half* Bt, const float* bias, const float* res,
                  void* out, int M, int N, int K, int flags) {
    if (N % 128 == 0) {
      dim3 g(N / 128, (M + 127) / 128);
      gemm_kernel<128><<<g, 256, 0, stream>>>(A, Bt, bias, res, out, M, N, K, flags);
    } else {
      dim3 g(N / 64, (M + 127) / 128);
      gemm_kernel<64><<<g, 256, 0, stream>>>(A, Bt, bias, res, out, M, N, K, flags);
    }
  };

  // ---- weight conversion (f32 -> f16, transposed to [N][K]) -------------
  k_convert<<<blocks((size_t)768 * 768), 256, 0, stream>>>(patch_w, patchW16, (size_t)768 * 768);
  for (int l = 0; l < 12; l++) {
    k_transpose_convert<<<blocks((size_t)2304 * 768), 256, 0, stream>>>(
        qkv_w + (size_t)l * 768 * 2304, qkvT + (size_t)l * 2304 * 768, 768, 2304);
    k_transpose_convert<<<blocks((size_t)768 * 768), 256, 0, stream>>>(
        proj_w + (size_t)l * 768 * 768, projT + (size_t)l * 768 * 768, 768, 768);
    k_transpose_convert<<<blocks((size_t)3072 * 768), 256, 0, stream>>>(
        fc1_w + (size_t)l * 768 * 3072, fc1T + (size_t)l * 3072 * 768, 768, 3072);
    k_transpose_convert<<<blocks((size_t)768 * 3072), 256, 0, stream>>>(
        fc2_w + (size_t)l * 3072 * 768, fc2T + (size_t)l * 768 * 3072, 3072, 768);
  }
  for (int f = 0; f < 2; f++) {
    k_transpose_convert<<<blocks((size_t)768 * 768), 256, 0, stream>>>(
        fsm_in_w + (size_t)f * 768 * 768, fsmT + (size_t)f * 768 * 768, 768, 768);
    k_transpose_convert<<<blocks((size_t)384 * 1152), 256, 0, stream>>>(
        o1_w + (size_t)f * 1152 * 384, o1T + (size_t)f * 384 * 1152, 1152, 384);
    k_transpose_convert<<<blocks((size_t)192 * 384), 256, 0, stream>>>(
        o2_w + (size_t)f * 384 * 192, o2T + (size_t)f * 192 * 384, 384, 192);
  }

  // ---- patch embedding ---------------------------------------------------
  k_patch_gather<<<blocks((size_t)9216 * 768), 256, 0, stream>>>(x, patchA);
  gemm(patchA, patchW16, patch_b, nullptr, hbuf, 9216, 768, 768, /*bias*/ 1);
  k_assemble<<<blocks((size_t)16 * 577 * 768), 256, 0, stream>>>(hbuf, cls, pos, tok);
  k_fill1<<<blocks(9216), 256, 0, stream>>>(pm, 9216);

  int fi = 0, use_policy = 0;
  for (int l = 0; l < 12; l++) {
    if (l == 4 || l == 8) {
      // ---- FSM token-pruning module ------------------------------------
      ln_kernel<<<9216, 256, 0, stream>>>(tok, fsm_ln_s + fi * 768, fsm_ln_b + fi * 768,
                                          lnbuf, 1 | 2);
      gemm(lnbuf, fsmT + (size_t)fi * 768 * 768, fsm_in_b + fi * 768, nullptr, hfsm,
           9216, 768, 768, 1 | 2);                               // bias + gelu, f32 out
      k_glb<<<16, 384, 0, stream>>>(hfsm, pm, glb);
      k_feat<<<blocks((size_t)9216 * 1152), 256, 0, stream>>>(hfsm, glb, noise, featb);
      gemm(featb, o1T + (size_t)fi * 384 * 1152, o1_b + fi * 384, nullptr, z1b,
           9216, 384, 1152, 1 | 2 | 8);
      gemm(z1b, o2T + (size_t)fi * 192 * 384, o2_b + fi * 192, nullptr, z2b,
           9216, 192, 384, 1 | 2 | 8);
      k_o3<<<blocks(9216), 256, 0, stream>>>(z2b, o3_w + (size_t)fi * 384, o3_b + fi * 2,
                                             gnoise + (size_t)fi * 9216 * 2, pm);
      k_policy<<<blocks(9232), 256, 0, stream>>>(pm, pol);
      use_policy = 1;
      fi++;
    }
    // ---- attention block -----------------------------------------------
    ln_kernel<<<9232, 256, 0, stream>>>(tok, ln1_s + l * 768, ln1_b + l * 768, lnbuf, 1);
    gemm(lnbuf, qkvT + (size_t)l * 2304 * 768, nullptr, nullptr, qkvbuf,
         9232, 2304, 768, 8);                                    // f16 out
    attn_kernel<<<dim3(37, 12, 16), 256, 0, stream>>>(qkvbuf, pol, attnout, use_policy);
    gemm(attnout, projT + (size_t)l * 768 * 768, proj_b + l * 768, tok, hbuf,
         9232, 768, 768, 1 | 4);                                 // bias + residual -> h
    // ---- MLP block -------------------------------------------------------
    ln_kernel<<<9232, 256, 0, stream>>>(hbuf, ln2_s + l * 768, ln2_b + l * 768, lnbuf, 1);
    gemm(lnbuf, fc1T + (size_t)l * 3072 * 768, fc1_b + l * 3072, nullptr, fc1out,
         9232, 3072, 768, 1 | 2 | 8);                            // bias + gelu, f16
    gemm(fc1out, fc2T + (size_t)l * 768 * 3072, fc2_b + l * 768, hbuf, tok,
         9232, 768, 3072, 1 | 4);                                // bias + residual -> tok
  }

  // ---- final layernorm -> d_out (f32) ------------------------------------
  ln_kernel<<<9232, 256, 0, stream>>>(tok, norm_s, norm_b, d_out, 0);
}
